// EchoEA_woEL_51032801411198
// MI455X (gfx1250) — matrix operand order repaired and checked by reading, not compiled, not run on hardware
//
#include <hip/hip_runtime.h>
#include <hip/hip_bf16.h>
#include <cmath>

#define HDIM 300
#define KPAD 320
#define OPAD 304
#define NKC  (KPAD / 32)   // 10 K-chunks of 32

typedef __attribute__((ext_vector_type(16))) __bf16 v16bf;
typedef __attribute__((ext_vector_type(8)))  float  v8f;

union FragBF { v16bf v; unsigned u[8]; };

__device__ __forceinline__ float waveReduceSum(float v) {
#pragma unroll
  for (int off = 16; off > 0; off >>= 1) v += __shfl_xor(v, off, 32);
  return v;
}

__device__ __forceinline__ void atomicMaxF32(float* addr, float val) {
  if (val >= 0.0f) atomicMax((int*)addr, __float_as_int(val));
  else             atomicMin((unsigned int*)addr, __float_as_uint(val));
}

// ---------------- elementwise helpers ----------------

__global__ void k_fill2d(float* __restrict__ p, int ld, int rows, float v) {
  long i = (long)blockIdx.x * blockDim.x + threadIdx.x;
  long total = (long)rows * HDIM;
  if (i >= total) return;
  int r = (int)(i / HDIM), c = (int)(i - (long)r * HDIM);
  p[(size_t)r * ld + c] = v;
}

__global__ void k_relu2d(float* __restrict__ p, int ld, int rows) {
  long i = (long)blockIdx.x * blockDim.x + threadIdx.x;
  long total = (long)rows * HDIM;
  if (i >= total) return;
  int r = (int)(i / HDIM), c = (int)(i - (long)r * HDIM);
  float v = p[(size_t)r * ld + c];
  p[(size_t)r * ld + c] = v > 0.0f ? v : 0.0f;
}

__global__ void k_fill1d(float* __restrict__ p, float v, int n) {
  int i = blockIdx.x * blockDim.x + threadIdx.x;
  if (i < n) p[i] = v;
}

// ---------------- l2 normalize (wave per row) ----------------

__global__ void k_l2norm(const float* __restrict__ x, float* __restrict__ y, int n) {
  int wave = (int)(((long)blockIdx.x * blockDim.x + threadIdx.x) >> 5);
  int lane = threadIdx.x & 31;
  if (wave >= n) return;
  const float* row = x + (size_t)wave * HDIM;
  float ss = 0.0f;
  for (int k = lane; k < HDIM; k += 32) { float v = row[k]; ss += v * v; }
  ss = waveReduceSum(ss);
  float inv = 1.0f / fmaxf(sqrtf(ss), 1e-12f);
  float* out = y + (size_t)wave * HDIM;
  for (int k = lane; k < HDIM; k += 32) out[k] = row[k] * inv;
}

// ---------------- weight fp32 -> padded bf16 [OPAD x KPAD] ----------------

__global__ void k_convW(const float* __restrict__ W, __hip_bfloat16* __restrict__ Wb) {
  int idx = blockIdx.x * blockDim.x + threadIdx.x;
  if (idx >= OPAD * KPAD) return;
  int o = idx / KPAD, k = idx - o * KPAD;
  float v = (o < HDIM && k < HDIM) ? W[o * HDIM + k] : 0.0f;
  Wb[idx] = __float2bfloat16(v);
}

// ---------------- bf16 WMMA GEMM: Y[r, o] = sum_k X[r,k] * W[o,k] ----------------
// Block = 16 output rows, 8 waves sharing an LDS-staged bf16 A tile.
// Each wave runs TWO independent WMMA chains (col tiles w and w+8) with
// double-buffered B fragments; waves 0..2 then do the tail tile w+16.

__global__ void __launch_bounds__(256)
k_gemm_bf16(const float* __restrict__ X, int ldx,
            const __hip_bfloat16* __restrict__ Wb,
            float* __restrict__ Y, int ldy, int nrows) {
  __shared__ __hip_bfloat16 As[16 * KPAD];
  int rowbase = blockIdx.x * 16;
  for (int idx = threadIdx.x; idx < 16 * KPAD; idx += 256) {
    int r = idx / KPAD, k = idx - r * KPAD;
    int row = rowbase + r;
    float v = (k < HDIM && row < nrows) ? X[(size_t)row * ldx + k] : 0.0f;
    As[idx] = __float2bfloat16(v);
  }
  __syncthreads();

  int wave = threadIdx.x >> 5;
  int lane = threadIdx.x & 31;
  int half = lane >> 4;    // 0: lanes 0-15, 1: lanes 16-31
  int mrow = lane & 15;    // M row for A frag / N col for B frag

  // A fragments: all 10 chunks from LDS once, reused for every column tile.
  FragBF a[NKC];
#pragma unroll
  for (int kc = 0; kc < NKC; ++kc) {
    int k0 = kc * 32;
#pragma unroll
    for (int i = 0; i < 8; ++i) {
      int ka = k0 + ((i < 4) ? (half * 8 + 2 * i) : (16 + half * 8 + 2 * (i - 4)));
      a[kc].u[i] = *(const unsigned*)&As[mrow * KPAD + ka];
    }
  }

  auto loadB = [&](const __hip_bfloat16* wr, int kc, FragBF& b) {
    int k0 = kc * 32 + half * 8;
#pragma unroll
    for (int i = 0; i < 8; ++i)
      b.u[i] = *(const unsigned*)&wr[k0 + ((i < 4) ? 2 * i : 16 + 2 * (i - 4))];
  };

  auto storeC = [&](const v8f& acc, int t) {
#pragma unroll
    for (int r = 0; r < 8; ++r) {
      int row = rowbase + r + half * 8;   // C/D layout: VGPR r -> M=r / M=r+8
      int col = t * 16 + mrow;
      if (row < nrows && col < HDIM) Y[(size_t)row * ldy + col] = acc[r];
    }
  };

  // --- dual-tile main pass: tiles t0 = wave, t1 = wave + 8 (both < 19) ---
  {
    int t0 = wave, t1 = wave + 8;
    const __hip_bfloat16* w0 = Wb + (size_t)(t0 * 16 + mrow) * KPAD;
    const __hip_bfloat16* w1 = Wb + (size_t)(t1 * 16 + mrow) * KPAD;
    v8f acc0 = {}, acc1 = {};
    FragBF b0a, b0b, b1a, b1b;
    loadB(w0, 0, b0a);
    loadB(w1, 0, b1a);
#pragma unroll
    for (int kc = 0; kc < NKC; ++kc) {
      FragBF& c0 = (kc & 1) ? b0b : b0a;
      FragBF& c1 = (kc & 1) ? b1b : b1a;
      FragBF& n0 = (kc & 1) ? b0a : b0b;
      FragBF& n1 = (kc & 1) ? b1a : b1b;
      if (kc + 1 < NKC) { loadB(w0, kc + 1, n0); loadB(w1, kc + 1, n1); }
      acc0 = __builtin_amdgcn_wmma_f32_16x16x32_bf16(
          false, a[kc].v, false, c0.v, (short)0, acc0, false, false);
      acc1 = __builtin_amdgcn_wmma_f32_16x16x32_bf16(
          false, a[kc].v, false, c1.v, (short)0, acc1, false, false);
    }
    storeC(acc0, t0);
    storeC(acc1, t1);
  }

  // --- tail tile t2 = wave + 16 for waves 0..2 (wave-uniform branch) ---
  if (wave < 3) {
    int t2 = wave + 16;
    const __hip_bfloat16* w2 = Wb + (size_t)(t2 * 16 + mrow) * KPAD;
    v8f acc = {};
    FragBF ba, bb;
    loadB(w2, 0, ba);
#pragma unroll
    for (int kc = 0; kc < NKC; ++kc) {
      FragBF& cur = (kc & 1) ? bb : ba;
      FragBF& nxt = (kc & 1) ? ba : bb;
      if (kc + 1 < NKC) loadB(w2, kc + 1, nxt);
      acc = __builtin_amdgcn_wmma_f32_16x16x32_bf16(
          false, a[kc].v, false, cur.v, (short)0, acc, false, false);
    }
    storeC(acc, t2);
  }
}

// ---------------- GAT pieces ----------------

__global__ void k_node_dots(const float* __restrict__ x, int ldx,
                            const float* __restrict__ ai, const float* __restrict__ aj,
                            float* __restrict__ si, float* __restrict__ sj, int n) {
  int wave = (int)(((long)blockIdx.x * blockDim.x + threadIdx.x) >> 5);
  int lane = threadIdx.x & 31;
  if (wave >= n) return;
  const float* row = x + (size_t)wave * ldx;
  float a = 0.0f, b = 0.0f;
  for (int k = lane; k < HDIM; k += 32) { float v = row[k]; a += v * ai[k]; b += v * aj[k]; }
  a = waveReduceSum(a);
  b = waveReduceSum(b);
  if (lane == 0) { si[wave] = a; sj[wave] = b; }
}

__global__ void k_edge_max(const float* __restrict__ si, const float* __restrict__ sj,
                           const int* __restrict__ esrc, const int* __restrict__ edst,
                           float* __restrict__ evals, float* __restrict__ m, int E) {
  int e = blockIdx.x * blockDim.x + threadIdx.x;
  if (e >= E) return;
  float v = si[edst[e]] + sj[esrc[e]];
  v = v > 0.0f ? v : 0.01f * v;          // leaky_relu(0.01)
  evals[e] = v;
  atomicMaxF32(&m[edst[e]], v);
}

__global__ void k_edge_expsum(float* __restrict__ evals, const int* __restrict__ edst,
                              const float* __restrict__ m, float* __restrict__ denom, int E) {
  int e = blockIdx.x * blockDim.x + threadIdx.x;
  if (e >= E) return;
  int d = edst[e];
  float ex = expf(evals[e] - m[d]);
  evals[e] = ex;
  atomicAdd(&denom[d], ex);
}

// wave per edge; float4 gathers (HDIM % 4 == 0, all row strides 16B-aligned)
__global__ void k_gat_scatter(const float* __restrict__ evals, const float* __restrict__ denom,
                              const int* __restrict__ esrc, const int* __restrict__ edst,
                              const float* __restrict__ x, int ldx,
                              float* __restrict__ out, int ldo, int E) {
  int wave = (int)(((long)blockIdx.x * blockDim.x + threadIdx.x) >> 5);
  int lane = threadIdx.x & 31;
  if (wave >= E) return;
  int s = esrc[wave], d = edst[wave];
  float alpha = evals[wave] / (denom[d] + 1e-16f);
  const float4* xs4 = (const float4*)(x + (size_t)s * ldx);
  float* od = out + (size_t)d * ldo;
  for (int q = lane; q < HDIM / 4; q += 32) {
    float4 v = xs4[q];
    atomicAdd(&od[4 * q + 0], alpha * v.x);
    atomicAdd(&od[4 * q + 1], alpha * v.y);
    atomicAdd(&od[4 * q + 2], alpha * v.z);
    atomicAdd(&od[4 * q + 3], alpha * v.w);
  }
}

// ---------------- GCN pieces ----------------

__global__ void k_deg(const int* __restrict__ edst, float* __restrict__ deg, int E) {
  int e = blockIdx.x * blockDim.x + threadIdx.x;
  if (e < E) atomicAdd(&deg[edst[e]], 1.0f);
}

__global__ void k_dinv(float* __restrict__ deg, int n) {
  int i = blockIdx.x * blockDim.x + threadIdx.x;
  if (i >= n) return;
  float d = deg[i];
  deg[i] = d > 0.0f ? rsqrtf(d) : 0.0f;
}

__global__ void k_gcn_scatter(const float* __restrict__ dinv,
                              const int* __restrict__ esrc, const int* __restrict__ edst,
                              const float* __restrict__ h, int ldh,
                              float* __restrict__ out, int ldo, int E) {
  int wave = (int)(((long)blockIdx.x * blockDim.x + threadIdx.x) >> 5);
  int lane = threadIdx.x & 31;
  if (wave >= E) return;
  int s = esrc[wave], d = edst[wave];
  float w = dinv[s] * dinv[d];
  const float4* hs4 = (const float4*)(h + (size_t)s * ldh);
  float* od = out + (size_t)d * ldo;
  for (int q = lane; q < HDIM / 4; q += 32) {
    float4 v = hs4[q];
    atomicAdd(&od[4 * q + 0], w * v.x);
    atomicAdd(&od[4 * q + 1], w * v.y);
    atomicAdd(&od[4 * q + 2], w * v.z);
    atomicAdd(&od[4 * q + 3], w * v.w);
  }
}

// ---------------- highway epilogue: out = sig(gpre+b)*relu(agg) + (1-sig)*x1 ----------------

__global__ void k_highway(const float* __restrict__ x1, int ld1,
                          const float* __restrict__ agg, int lda,
                          const float* __restrict__ gpre, int ldg,
                          const float* __restrict__ bias,
                          float* __restrict__ out, int ldo, int n) {
  long i = (long)blockIdx.x * blockDim.x + threadIdx.x;
  long total = (long)n * HDIM;
  if (i >= total) return;
  int r = (int)(i / HDIM), c = (int)(i - (long)r * HDIM);
  float g = 1.0f / (1.0f + expf(-(gpre[(size_t)r * ldg + c] + bias[c])));
  float a = fmaxf(agg[(size_t)r * lda + c], 0.0f);
  float x = x1[(size_t)r * ld1 + c];
  out[(size_t)r * ldo + c] = g * a + (1.0f - g) * x;
}

// ---------------- launcher ----------------

extern "C" void kernel_launch(void* const* d_in, const int* in_sizes, int n_in,
                              void* d_out, int out_size, void* d_ws, size_t ws_size,
                              hipStream_t stream) {
  const float* x_e     = (const float*)d_in[0];
  const int*   eia     = (const int*)d_in[3];
  const float* gcn1_W  = (const float*)d_in[5];
  const float* hw1_W   = (const float*)d_in[6];
  const float* hw1_b   = (const float*)d_in[7];
  const float* gat1_ai = (const float*)d_in[8];
  const float* gat1_aj = (const float*)d_in[9];
  const float* ghw1_W  = (const float*)d_in[10];
  const float* ghw1_b  = (const float*)d_in[11];
  const float* gat2_ai = (const float*)d_in[12];
  const float* gat2_aj = (const float*)d_in[13];
  const float* ghw2_W  = (const float*)d_in[14];
  const float* ghw2_b  = (const float*)d_in[15];
  const float* gat_ai  = (const float*)d_in[16];
  const float* gat_aj  = (const float*)d_in[17];

  const int N = in_sizes[0] / HDIM;
  const int E = in_sizes[3] / 2;
  const int* esrc = eia;       // ei_j
  const int* edst = eia + E;   // ei_i
  float* dout = (float*)d_out; // [N x 600]

  // workspace carve-up
  size_t NH = (size_t)N * HDIM;
  float* bufX0  = (float*)d_ws;          // x0, later x3
  float* bufX2  = bufX0 + NH;
  float* bufH   = bufX2 + NH;
  float* bufG   = bufH + NH;
  float* bufAgg = bufG + NH;
  float* s_i    = bufAgg + NH;
  float* s_j    = s_i + N;
  float* mbuf   = s_j + N;
  float* dnm    = mbuf + N;
  float* ebuf   = dnm + N;
  __hip_bfloat16* wb0 = (__hip_bfloat16*)(ebuf + E);
  __hip_bfloat16* wb1 = wb0 + OPAD * KPAD;
  __hip_bfloat16* wb2 = wb1 + OPAD * KPAD;
  __hip_bfloat16* wb3 = wb2 + OPAD * KPAD;

  auto cdiv = [](long a, long b) { return (int)((a + b - 1) / b); };
  const int BT = 256;
  int gNH   = cdiv(NH, BT);             // elementwise over N*H
  int gNw   = cdiv((long)N * 32, BT);   // wave per node
  int gEw   = cdiv((long)E * 32, BT);   // wave per edge
  int gE    = cdiv(E, BT);
  int gN1   = cdiv(N, BT);
  int gW    = cdiv(OPAD * KPAD, BT);
  int gGemm = cdiv(N, 16);

  // weights -> bf16
  k_convW<<<gW, BT, 0, stream>>>(gcn1_W, wb0);
  k_convW<<<gW, BT, 0, stream>>>(hw1_W,  wb1);
  k_convW<<<gW, BT, 0, stream>>>(ghw1_W, wb2);
  k_convW<<<gW, BT, 0, stream>>>(ghw2_W, wb3);

  // x0 = l2_normalize(x_e)
  k_l2norm<<<gNw, BT, 0, stream>>>(x_e, bufX0, N);

  auto run_gat = [&](const float* x, int ldx, const float* ai, const float* aj,
                     float* out, int ldo) {
    k_node_dots<<<gNw, BT, 0, stream>>>(x, ldx, ai, aj, s_i, s_j, N);
    k_fill1d<<<gN1, BT, 0, stream>>>(mbuf, -INFINITY, N);
    k_fill1d<<<gN1, BT, 0, stream>>>(dnm, 0.0f, N);
    k_edge_max<<<gE, BT, 0, stream>>>(s_i, s_j, esrc, edst, ebuf, mbuf, E);
    k_edge_expsum<<<gE, BT, 0, stream>>>(ebuf, edst, mbuf, dnm, E);
    k_fill2d<<<gNH, BT, 0, stream>>>(out, ldo, N, 0.0f);
    k_gat_scatter<<<gEw, BT, 0, stream>>>(ebuf, dnm, esrc, edst, x, ldx, out, ldo, E);
  };

  // ---- layer 1: x2 = highway(x0, relu(GCN(x0)), hw1) ----
  k_gemm_bf16<<<gGemm, BT, 0, stream>>>(bufX0, HDIM, wb0, bufH, HDIM, N);  // h = x0 @ gcn1_W.T
  k_fill1d<<<gN1, BT, 0, stream>>>(dnm, 0.0f, N);
  k_deg<<<gE, BT, 0, stream>>>(edst, dnm, E);
  k_dinv<<<gN1, BT, 0, stream>>>(dnm, N);
  k_fill2d<<<gNH, BT, 0, stream>>>(bufAgg, HDIM, N, 0.0f);
  k_gcn_scatter<<<gEw, BT, 0, stream>>>(dnm, esrc, edst, bufH, HDIM, bufAgg, HDIM, E);
  k_gemm_bf16<<<gGemm, BT, 0, stream>>>(bufX0, HDIM, wb1, bufG, HDIM, N);  // gate pre
  k_highway<<<gNH, BT, 0, stream>>>(bufX0, HDIM, bufAgg, HDIM, bufG, HDIM, hw1_b,
                                    bufX2, HDIM, N);

  // ---- layer 2: x3 = highway(x2, relu(GAT(x2)), ghw1)  (x3 stored in bufX0) ----
  run_gat(bufX2, HDIM, gat1_ai, gat1_aj, bufAgg, HDIM);
  k_gemm_bf16<<<gGemm, BT, 0, stream>>>(bufX2, HDIM, wb2, bufG, HDIM, N);
  k_highway<<<gNH, BT, 0, stream>>>(bufX2, HDIM, bufAgg, HDIM, bufG, HDIM, ghw1_b,
                                    bufX0, HDIM, N);

  // ---- layer 3: xe = highway(x2, relu(GAT(x3)), ghw2) -> d_out[:, 0:300] ----
  run_gat(bufX0, HDIM, gat2_ai, gat2_aj, bufAgg, HDIM);
  k_gemm_bf16<<<gGemm, BT, 0, stream>>>(bufX2, HDIM, wb3, bufG, HDIM, N);
  k_highway<<<gNH, BT, 0, stream>>>(bufX2, HDIM, bufAgg, HDIM, bufG, HDIM, ghw2_b,
                                    dout, 2 * HDIM, N);

  // ---- layer 4: ye = relu-GAT(xe) -> d_out[:, 300:600] ----
  run_gat(dout, 2 * HDIM, gat_ai, gat_aj, dout + HDIM, 2 * HDIM);
  k_relu2d<<<gNH, BT, 0, stream>>>(dout + HDIM, 2 * HDIM, N);
}
